// GCN_41686952575411
// MI455X (gfx1250) — matrix operand order repaired and checked by reading, not compiled, not run on hardware
//
#include <hip/hip_runtime.h>
#include <hip/hip_bf16.h>
#include <math.h>

#define N_NODES 8192
#define N_EDGES 262144

typedef __attribute__((ext_vector_type(16))) __bf16 v16bf;
typedef __attribute__((ext_vector_type(8)))  __bf16 v8bf;
typedef __attribute__((ext_vector_type(8)))  float  v8f;

// ---------------------------------------------------------------- utilities
__global__ void k_fill_f32(float* p, float v, int n) {
  int i = blockIdx.x * blockDim.x + threadIdx.x;
  if (i < n) p[i] = v;
}
__global__ void k_fill_i32(int* p, int v, int n) {
  int i = blockIdx.x * blockDim.x + threadIdx.x;
  if (i < n) p[i] = v;
}
// f32 -> bf16 bulk convert (n multiple of 4)
__global__ void k_cvt_bf16(const float* __restrict__ s, __bf16* __restrict__ d, int n) {
  int i = (blockIdx.x * blockDim.x + threadIdx.x) * 4;
  if (i < n) {
    float4 v = *reinterpret_cast<const float4*>(s + i);
    d[i]     = (__bf16)v.x;
    d[i + 1] = (__bf16)v.y;
    d[i + 2] = (__bf16)v.z;
    d[i + 3] = (__bf16)v.w;
  }
}

__device__ inline void atomicMaxFloat(float* addr, float val) {
  int old = __float_as_int(*addr);
  while (__int_as_float(old) < val) {
    int assumed = old;
    old = atomicCAS((int*)addr, assumed, __float_as_int(val));
    if (old == assumed) break;
  }
}

// ---------------------------------------------------------------- degrees / norms
__global__ void k_deg(const int* __restrict__ ei, float* __restrict__ deg, int E) {
  int e = blockIdx.x * blockDim.x + threadIdx.x;
  if (e < E) atomicAdd(&deg[ei[E + e]], 1.0f);
}
__global__ void k_dinv(const float* __restrict__ deg, float* __restrict__ dinv, int n) {
  int i = blockIdx.x * blockDim.x + threadIdx.x;
  if (i < n) dinv[i] = rsqrtf(deg[i]);   // deg >= 1 (self loop)
}

// ---------------------------------------------------------------- CSR build (group edges by row)
__global__ void k_rowcnt(const int* __restrict__ ei, int* __restrict__ cnt, int E) {
  int e = blockIdx.x * blockDim.x + threadIdx.x;
  if (e < E) atomicAdd(&cnt[ei[e]], 1);
}
// single block, 1024 threads, 8 elements each -> exclusive scan of 8192 counts
__global__ void k_scan(const int* __restrict__ cnt, int* __restrict__ rowstart, int n) {
  __shared__ int lds[1024];
  int t = threadIdx.x;
  int base = t * 8;
  int local[8];
  int s = 0;
#pragma unroll
  for (int i = 0; i < 8; ++i) { local[i] = s; s += cnt[base + i]; }
  lds[t] = s;
  __syncthreads();
  for (int off = 1; off < 1024; off <<= 1) {
    int v = (t >= off) ? lds[t - off] : 0;
    __syncthreads();
    lds[t] += v;
    __syncthreads();
  }
  int pre = (t == 0) ? 0 : lds[t - 1];
#pragma unroll
  for (int i = 0; i < 8; ++i) rowstart[base + i] = pre + local[i];
  if (t == 1023) rowstart[n] = lds[1023];
}
__global__ void k_csrfill(const int* __restrict__ ei, const int* __restrict__ rowstart,
                          int* __restrict__ cursor, int* __restrict__ csr, int E) {
  int e = blockIdx.x * blockDim.x + threadIdx.x;
  if (e < E) {
    int r = ei[e];
    int p = atomicAdd(&cursor[r], 1);
    csr[rowstart[r] + p] = ei[E + e];
  }
}

// ---------------------------------------------------------------- bf16 WMMA GEMM
// C[M,Nn] = A[M,K] @ W[K,Nn] (+bias)(+leaky).  A is pre-converted bf16.
// Block = 256 thr (8 waves); block owns one 16-col N-tile; W slice staged in LDS (bf16).
// Each wave computes MT=4 consecutive 16x16 M-tiles, reusing one B fragment per k-step.
// K is compile-time -> fully unrolled: K/32 * MT static v_wmma sites per kernel.
template <int K, int MT>
__global__ void k_wmma_gemm(const __bf16* __restrict__ A, const float* __restrict__ W,
                            const float* __restrict__ bias, float* __restrict__ C,
                            int M, int Nn, int flags /*1=bias,2=leaky*/) {
  constexpr int LSTRIDE = K + 16;            // elements; keeps 32B alignment, 8-bank skew
  __shared__ __bf16 ldsW[16 * LSTRIDE];

  const int tileN = blockIdx.x;
  const int ncol0 = tileN * 16;

  // cooperative stage of W[:, ncol0..ncol0+15] into LDS (column-major, bf16)
  for (int idx = threadIdx.x; idx < 16 * K; idx += 256) {
    int col = idx & 15;
    int k   = idx >> 4;
    ldsW[col * LSTRIDE + k] = (__bf16)W[(size_t)k * Nn + ncol0 + col];
  }
  __syncthreads();

  const int lane  = threadIdx.x & 31;
  const int khalf = lane >> 4;                       // A/B K-half select
  const int waveM = blockIdx.y * (8 * MT * 16) + (threadIdx.x >> 5) * (MT * 16);
  const int ncol  = ncol0 + (lane & 15);
  const __bf16* ldsCol = ldsW + (size_t)(lane & 15) * LSTRIDE;

  v8f acc[MT];
#pragma unroll
  for (int mt = 0; mt < MT; ++mt) acc[mt] = (v8f){};

#pragma unroll
  for (int k0 = 0; k0 < K; k0 += 32) {
    // B fragment 32x16 from LDS: lanes 0-15 K=k0..+15, lanes 16-31 K=k0+16..+31
    v16bf b = *reinterpret_cast<const v16bf*>(ldsCol + k0 + khalf * 16);
#pragma unroll
    for (int mt = 0; mt < MT; ++mt) {
      const __bf16* Ar = A + (size_t)(waveM + mt * 16 + (lane & 15)) * K + k0;
      v8bf lo = *reinterpret_cast<const v8bf*>(Ar + khalf * 16);
      v8bf hi = *reinterpret_cast<const v8bf*>(Ar + khalf * 16 + 8);
      v16bf a;
#pragma unroll
      for (int i = 0; i < 8; ++i) { a[i] = lo[i]; a[8 + i] = hi[i]; }
      acc[mt] = __builtin_amdgcn_wmma_f32_16x16x32_bf16(false, a, false, b,
                                                        (short)0, acc[mt], false, false);
    }
  }

  float bv = (flags & 1) ? bias[ncol] : 0.0f;
#pragma unroll
  for (int mt = 0; mt < MT; ++mt) {
#pragma unroll
    for (int r = 0; r < 8; ++r) {
      int mr  = waveM + mt * 16 + ((lane < 16) ? r : r + 8);
      float v = acc[mt][r] + bv;
      if (flags & 2) v = (v > 0.0f) ? v : 0.01f * v;
      C[(size_t)mr * Nn + ncol] = v;
    }
  }
}

// ---------------------------------------------------------------- GCN aggregation
__global__ void k_scatter(const int* __restrict__ ei, const float* __restrict__ dinv,
                          const float* __restrict__ xw, float* __restrict__ out,
                          int E, int F) {
  int e = blockIdx.x, f = threadIdx.x;
  int r = ei[e], c = ei[E + e];
  float nrm = dinv[r] * dinv[c];
  atomicAdd(&out[(size_t)c * F + f], nrm * xw[(size_t)r * F + f]);
}
__global__ void k_gcn_final(const float* __restrict__ xw, const float* __restrict__ dinv,
                            const float* __restrict__ b, float* __restrict__ out,
                            int F, int leaky) {
  int n = blockIdx.x, f = threadIdx.x;
  float d = dinv[n];
  float v = out[(size_t)n * F + f] + d * d * xw[(size_t)n * F + f] + b[f];
  if (leaky) v = (v > 0.0f) ? v : 0.01f * v;
  out[(size_t)n * F + f] = v;
}

// ---------------------------------------------------------------- attention helpers
__global__ void k_colsum(const float* __restrict__ H, float* __restrict__ cs, int n) {
  int f = threadIdx.x;  // 256
  float s = 0.0f;
  for (int i = 0; i < n; ++i) s += H[(size_t)i * 256 + f];
  cs[f] = s;
}

// sparse closed-form of softmax(QK^T * mask) @ H. One wave per node.
__global__ void k_attn(const float* __restrict__ Q, const float* __restrict__ Kt,
                       const float* __restrict__ H, const int* __restrict__ rowstart,
                       const int* __restrict__ csr, const float* __restrict__ colsum,
                       float* __restrict__ sbuf, float* __restrict__ O, int n) {
  int node = blockIdx.x * (blockDim.x >> 5) + (threadIdx.x >> 5);
  int lane = threadIdx.x & 31;
  if (node >= n) return;
  float q[8];
#pragma unroll
  for (int j = 0; j < 8; ++j) q[j] = Q[(size_t)node * 256 + j * 32 + lane];
  int s = rowstart[node], e = rowstart[node + 1];
  float m = 0.0f;                           // masked-out entries score 0
  for (int t = s; t < e; ++t) {
    int j = csr[t];
    if (t + 1 < e) __builtin_prefetch(Kt + (size_t)csr[t + 1] * 256 + lane, 0, 1);
    float p = 0.0f;
#pragma unroll
    for (int jj = 0; jj < 8; ++jj) p += q[jj] * Kt[(size_t)j * 256 + jj * 32 + lane];
#pragma unroll
    for (int off = 16; off > 0; off >>= 1) p += __shfl_xor(p, off, 32);
    if (lane == 0) sbuf[t] = p;
    m = fmaxf(m, p);
  }
  float acc[8] = {}, hs[8] = {};
  float Z = 0.0f;
  for (int t = s; t < e; ++t) {
    int j = csr[t];
    if (t + 1 < e) __builtin_prefetch(H + (size_t)csr[t + 1] * 256 + lane, 0, 1);
    float w = __expf(sbuf[t] - m);
    Z += w;
#pragma unroll
    for (int jj = 0; jj < 8; ++jj) {
      float hv = H[(size_t)j * 256 + jj * 32 + lane];
      acc[jj] += w * hv;
      hs[jj]  += hv;
    }
  }
  float base = __expf(-m);
  Z += base * (float)(n - (e - s));
  float invZ = 1.0f / Z;
#pragma unroll
  for (int jj = 0; jj < 8; ++jj)
    O[(size_t)node * 256 + jj * 32 + lane] =
        (acc[jj] + base * (colsum[jj * 32 + lane] - hs[jj])) * invZ;
}

// ---------------------------------------------------------------- residual + layernorm + max-pool
__global__ void k_ln_pool(const float* __restrict__ h, const float* __restrict__ res,
                          const float* __restrict__ g, const float* __restrict__ b,
                          float* __restrict__ pooled, int n) {
  int node = blockIdx.x * (blockDim.x >> 5) + (threadIdx.x >> 5);
  int lane = threadIdx.x & 31;
  if (node >= n) return;
  float v0 = h[(size_t)node * 64 + lane]      + res[(size_t)node * 64 + lane];
  float v1 = h[(size_t)node * 64 + 32 + lane] + res[(size_t)node * 64 + 32 + lane];
  float s = v0 + v1;
#pragma unroll
  for (int off = 16; off > 0; off >>= 1) s += __shfl_xor(s, off, 32);
  float mu = s * (1.0f / 64.0f);
  float d0 = v0 - mu, d1 = v1 - mu;
  float vs = d0 * d0 + d1 * d1;
#pragma unroll
  for (int off = 16; off > 0; off >>= 1) vs += __shfl_xor(vs, off, 32);
  float rstd = rsqrtf(vs * (1.0f / 64.0f) + 1e-5f);
  atomicMaxFloat(&pooled[lane],      d0 * rstd * g[lane]      + b[lane]);
  atomicMaxFloat(&pooled[32 + lane], d1 * rstd * g[32 + lane] + b[32 + lane]);
}

__global__ void k_final(const float* __restrict__ pooled, const float* __restrict__ Wf,
                        const float* __restrict__ bf, float* __restrict__ out) {
  int o = threadIdx.x;
  if (o >= 16) return;
  float s = bf[o];
  for (int f = 0; f < 64; ++f) s += pooled[f] * Wf[f * 16 + o];
  out[o] = s;
}

// ---------------------------------------------------------------- host launcher
extern "C" void kernel_launch(void* const* d_in, const int* in_sizes, int n_in,
                              void* d_out, int out_size, void* d_ws, size_t ws_size,
                              hipStream_t stream) {
  const float* x    = (const float*)d_in[0];
  const int*   ei   = (const int*)d_in[1];
  const float* W1   = (const float*)d_in[2];
  const float* b1   = (const float*)d_in[3];
  const float* W2   = (const float*)d_in[4];
  const float* b2   = (const float*)d_in[5];
  const float* W3   = (const float*)d_in[6];
  const float* b3   = (const float*)d_in[7];
  const float* Wk   = (const float*)d_in[8];
  const float* bk   = (const float*)d_in[9];
  const float* Wq   = (const float*)d_in[10];
  const float* bq   = (const float*)d_in[11];
  const float* Wr   = (const float*)d_in[12];
  const float* br   = (const float*)d_in[13];
  const float* ln_g = (const float*)d_in[14];
  const float* ln_b = (const float*)d_in[15];
  const float* Wf   = (const float*)d_in[16];
  const float* bf   = (const float*)d_in[17];
  float* out = (float*)d_out;

  const int N = N_NODES, E = N_EDGES;

  // workspace bump allocator (256B aligned)
  char* ws = (char*)d_ws;
  size_t cur = 0;
  auto alloc = [&](size_t bytes) {
    void* p = ws + cur;
    cur = (cur + bytes + 255) & ~(size_t)255;
    return p;
  };
  float*  deg      = (float*)alloc(sizeof(float) * N);
  float*  dinv     = (float*)alloc(sizeof(float) * N);
  int*    rowcnt   = (int*)alloc(sizeof(int) * N);
  int*    rowstart = (int*)alloc(sizeof(int) * (N + 1));
  int*    cursor   = (int*)alloc(sizeof(int) * N);
  int*    csr      = (int*)alloc(sizeof(int) * E);
  float*  sbuf     = (float*)alloc(sizeof(float) * E);
  float*  xw       = (float*)alloc(sizeof(float) * N * 256);
  float*  h1       = (float*)alloc(sizeof(float) * N * 128);
  float*  h2       = (float*)alloc(sizeof(float) * N * 256);
  float*  keyb     = (float*)alloc(sizeof(float) * N * 256);
  float*  qryb     = (float*)alloc(sizeof(float) * N * 256);
  float*  h3       = (float*)alloc(sizeof(float) * N * 256);
  float*  h4       = (float*)alloc(sizeof(float) * N * 64);
  float*  resb     = (float*)alloc(sizeof(float) * N * 64);
  float*  csum     = (float*)alloc(sizeof(float) * 256);
  float*  pooled   = (float*)alloc(sizeof(float) * 64);
  __bf16* xb       = (__bf16*)alloc(sizeof(__bf16) * N * 128);
  __bf16* h1b      = (__bf16*)alloc(sizeof(__bf16) * N * 128);
  __bf16* h2b      = (__bf16*)alloc(sizeof(__bf16) * N * 256);
  __bf16* h3b      = (__bf16*)alloc(sizeof(__bf16) * N * 256);

  // degrees (self-loop baked in via init=1.0) and symmetric norms
  k_fill_f32<<<N / 256, 256, 0, stream>>>(deg, 1.0f, N);
  k_deg<<<E / 256, 256, 0, stream>>>(ei, deg, E);
  k_dinv<<<N / 256, 256, 0, stream>>>(deg, dinv, N);

  // CSR by attention row
  k_fill_i32<<<N / 256, 256, 0, stream>>>(rowcnt, 0, N);
  k_rowcnt<<<E / 256, 256, 0, stream>>>(ei, rowcnt, E);
  k_scan<<<1, 1024, 0, stream>>>(rowcnt, rowstart, N);
  k_fill_i32<<<N / 256, 256, 0, stream>>>(cursor, 0, N);
  k_csrfill<<<E / 256, 256, 0, stream>>>(ei, rowstart, cursor, csr, E);

  // bf16 image of x
  k_cvt_bf16<<<N * 128 / 4 / 256, 256, 0, stream>>>(x, xb, N * 128);

  // residual = x @ Wr + br   [8192,128]x[128,64]
  k_wmma_gemm<128, 4><<<dim3(64 / 16, 16), 256, 0, stream>>>(xb, Wr, br, resb, N, 64, 1);

  // GCN1: xw = x@W1 ; aggregate ; +self +b1 ; leaky
  k_wmma_gemm<128, 4><<<dim3(128 / 16, 16), 256, 0, stream>>>(xb, W1, nullptr, xw, N, 128, 0);
  k_fill_f32<<<N * 128 / 256, 256, 0, stream>>>(h1, 0.0f, N * 128);
  k_scatter<<<E, 128, 0, stream>>>(ei, dinv, xw, h1, E, 128);
  k_gcn_final<<<N, 128, 0, stream>>>(xw, dinv, b1, h1, 128, 1);

  // GCN2: [8192,128]x[128,256]
  k_cvt_bf16<<<N * 128 / 4 / 256, 256, 0, stream>>>(h1, h1b, N * 128);
  k_wmma_gemm<128, 4><<<dim3(256 / 16, 16), 256, 0, stream>>>(h1b, W2, nullptr, xw, N, 256, 0);
  k_fill_f32<<<N * 256 / 256, 256, 0, stream>>>(h2, 0.0f, N * 256);
  k_scatter<<<E, 256, 0, stream>>>(ei, dinv, xw, h2, E, 256);
  k_gcn_final<<<N, 256, 0, stream>>>(xw, dinv, b2, h2, 256, 1);

  // keys / queries  [8192,256]x[256,256] (+bias)
  k_cvt_bf16<<<N * 256 / 4 / 256, 256, 0, stream>>>(h2, h2b, N * 256);
  k_wmma_gemm<256, 4><<<dim3(256 / 16, 16), 256, 0, stream>>>(h2b, Wk, bk, keyb, N, 256, 1);
  k_wmma_gemm<256, 4><<<dim3(256 / 16, 16), 256, 0, stream>>>(h2b, Wq, bq, qryb, N, 256, 1);

  // sparse closed-form masked softmax attention
  k_colsum<<<1, 256, 0, stream>>>(h2, csum, N);
  k_attn<<<N / 8, 256, 0, stream>>>(qryb, keyb, h2, rowstart, csr, csum, sbuf, h3, N);

  // GCN3: [8192,256]x[256,64]
  k_cvt_bf16<<<N * 256 / 4 / 256, 256, 0, stream>>>(h3, h3b, N * 256);
  k_wmma_gemm<256, 4><<<dim3(64 / 16, 16), 256, 0, stream>>>(h3b, W3, nullptr, xw, N, 64, 0);
  k_fill_f32<<<N * 64 / 256, 256, 0, stream>>>(h4, 0.0f, N * 64);
  k_scatter<<<E, 64, 0, stream>>>(ei, dinv, xw, h4, E, 64);
  k_gcn_final<<<N, 64, 0, stream>>>(xw, dinv, b3, h4, 64, 1);

  // residual add + layernorm + global max pool + final linear
  k_fill_f32<<<1, 64, 0, stream>>>(pooled, -3.0e38f, 64);
  k_ln_pool<<<N / 8, 256, 0, stream>>>(h4, resb, ln_g, ln_b, pooled, N);
  k_final<<<1, 16, 0, stream>>>(pooled, Wf, bf, out);
}
